// LSTM_78649441125003
// MI455X (gfx1250) — matrix operand order repaired and checked by reading, not compiled, not run on hardware
//
#include <hip/hip_runtime.h>
#include <math.h>

// Problem constants (from reference): dim=512, T=512, B=64, out=1000
#define DIM    512
#define NROWS  2048        // 4*dim gate rows
#define KW     1024        // W row stride (dim + dim)
#define BT     32768       // B*T sequential LSTM steps
#define OUT_SZ 1000
#define BATCH  64
#define TLEN   512

typedef __attribute__((ext_vector_type(16))) __bf16 v16bf;
typedef __attribute__((ext_vector_type(8)))  float  v8f;
typedef __attribute__((ext_vector_type(4)))  unsigned int v4u;
typedef __attribute__((ext_vector_type(4)))  int v4i;
typedef __attribute__((ext_vector_type(8)))  int v8i;
typedef __attribute__((address_space(3))) unsigned short lds_ushort;

union Frag16 {
  unsigned int   ui[8];
  unsigned short us[16];
  v16bf          v;
};

__device__ __forceinline__ unsigned short f2bf(float f) {
  unsigned u = __float_as_uint(f);
  u += 0x7fffu + ((u >> 16) & 1u);          // round-to-nearest-even
  return (unsigned short)(u >> 16);
}
__device__ __forceinline__ float bf2f(unsigned short s) {
  return __uint_as_float(((unsigned)s) << 16);
}

// ---------------------------------------------------------------------------
// Phase 1: zx[BT x 2048] = X[BT x 512] @ Wx^T + b   (bf16 WMMA, f32 accum)
// Block tile 128(M) x 64(N), 8 waves, each wave a 32x32 (2x2 of 16x16) tile.
// ---------------------------------------------------------------------------
#define BM 128
#define BN 64
#define BK 32
#define LSTR 34   // LDS row stride in ushorts: 17 banks -> conflict free

__global__ __launch_bounds__(256) void zx_gemm_bf16_wmma(
    const float* __restrict__ X, const float* __restrict__ W,
    const float* __restrict__ bias, float* __restrict__ ZX) {
  __shared__ unsigned short As[BM * LSTR];
  __shared__ unsigned short Bs[BN * LSTR];

  const int tid  = threadIdx.x;
  const int lane = tid & 31;
  const int wave = tid >> 5;
  const int wm = wave & 3;          // 4 waves along M
  const int wn = wave >> 2;         // 2 waves along N
  const int m0 = blockIdx.y * BM;
  const int n0 = blockIdx.x * BN;
  const int half = lane >> 4;       // 0: lanes 0-15, 1: lanes 16-31
  const int lrow = lane & 15;

  v8f acc[2][2] = {};

  for (int kb = 0; kb < DIM / BK; ++kb) {
    // stage A (128x32) and B (64x32) tiles as bf16 in LDS
#pragma unroll
    for (int i = 0; i < (BM * BK) / 256; ++i) {
      int e = tid + i * 256, m = e >> 5, k = e & (BK - 1);
      As[m * LSTR + k] = f2bf(X[(size_t)(m0 + m) * DIM + kb * BK + k]);
    }
#pragma unroll
    for (int i = 0; i < (BN * BK) / 256; ++i) {
      int e = tid + i * 256, n = e >> 5, k = e & (BK - 1);
      Bs[n * LSTR + k] = f2bf(W[(size_t)(n0 + n) * KW + DIM + kb * BK + k]);
    }
    if (kb + 1 < DIM / BK)  // warm L2 for the next A tile (global_prefetch_b8)
      __builtin_prefetch(&X[(size_t)(m0 + (tid >> 1)) * DIM + (kb + 1) * BK +
                            ((tid & 1) << 4)], 0, 1);
    __syncthreads();

    // Build fragments in contiguous k order (k = 16*half + s), same for A and B
    Frag16 af[2], bf[2];
#pragma unroll
    for (int mi = 0; mi < 2; ++mi) {
      int m = wm * 32 + mi * 16 + lrow;
#pragma unroll
      for (int v = 0; v < 8; ++v) {
        int k0 = (half ? 16 : 0) + 2 * v;
        af[mi].ui[v] = *(const unsigned int*)&As[m * LSTR + k0];
      }
    }
#pragma unroll
    for (int ni = 0; ni < 2; ++ni) {
      int n = wn * 32 + ni * 16 + lrow;
      int kb2 = half ? 16 : 0;
#pragma unroll
      for (int v = 0; v < 8; ++v)
        bf[ni].ui[v] = *(const unsigned int*)&Bs[n * LSTR + kb2 + 2 * v];
    }
#pragma unroll
    for (int mi = 0; mi < 2; ++mi)
#pragma unroll
      for (int ni = 0; ni < 2; ++ni)
        acc[mi][ni] = __builtin_amdgcn_wmma_f32_16x16x32_bf16(
            false, af[mi].v, false, bf[ni].v, (short)0, acc[mi][ni],
            false, false);
    __syncthreads();
  }

  // Epilogue: D layout -> VGPR v holds M = v + 8*half, N = lrow; add bias
#pragma unroll
  for (int mi = 0; mi < 2; ++mi)
#pragma unroll
    for (int ni = 0; ni < 2; ++ni) {
      int gcol = n0 + wn * 32 + ni * 16 + lrow;
      float bv = bias[gcol];
#pragma unroll
      for (int v = 0; v < 8; ++v) {
        int grow = m0 + wm * 32 + mi * 16 + v + (half ? 8 : 0);
        ZX[(size_t)grow * NROWS + gcol] = acc[mi][ni][v] + bv;
      }
    }
}

// ---------------------------------------------------------------------------
// Wh pre-pass: f32 gate-interleaved -> block-major bf16 so each recurrence
// WG's slice is one contiguous 32 KB run the TDM can fetch in one descriptor.
// ---------------------------------------------------------------------------
#define RB  64      // persistent blocks
#define DPB 8       // dims per block
#define RPB 32      // gate rows per block
#define WSTR 514    // padded LDS row stride in ushorts (TDM pad produces this)

__global__ __launch_bounds__(256) void wh_to_bf16(
    const float* __restrict__ W, unsigned short* __restrict__ WHBF) {
  int e = blockIdx.x * 256 + threadIdx.x;   // 0 .. 2048*512-1
  int k = e & (DIM - 1);
  int j = (e >> 9) & (RPB - 1);
  int blk = e >> 14;
  int grow = (j >> 3) * DIM + blk * DPB + (j & 7);
  WHBF[e] = f2bf(W[(size_t)grow * KW + k]);
}

// Issue one TDM descriptor: 512x32 bf16 tile, global row stride 512,
// LDS padding +1 DWORD every 256 DWORDs -> LDS row stride 514 ushorts.
// D# layout per CDNA5 ISA.
__device__ __forceinline__ void tdm_load_wh(const unsigned short* gsrc,
                                            unsigned lds_byte_off) {
  unsigned long long ga = (unsigned long long)(size_t)gsrc;
  v4u g0;
  g0[0] = 1u;                                   // count=1 (valid user D#)
  g0[1] = lds_byte_off;                         // lds_addr
  g0[2] = (unsigned)(ga & 0xffffffffu);         // global_addr[31:0]
  g0[3] = (unsigned)((ga >> 32) & 0x01ffffffu)  // global_addr[56:32]
          | (2u << 30);                         // type=2 ("image")
  v8i g1;
  g1[0] = (int)((1u << 16)      // data_size=1 -> 2-byte elements
              | (1u << 20)      // pad_enable
              | (7u << 22));    // pad_interval=7 -> every 256 DWORDs (+1 DWORD)
  g1[1] = (int)(512u << 16);    // tensor_dim0[15:0] = 512
  g1[2] = (int)(32u << 16);     // tensor_dim0 hi = 0, tensor_dim1 = 32
  g1[3] = (int)(512u << 16);    // tensor_dim1 hi = 0, tile_dim0 = 512
  g1[4] = 32;                   // tile_dim1 = 32, tile_dim2 = 0
  g1[5] = 512;                  // tensor_dim0_stride[31:0] = 512
  g1[6] = 0;                    // stride hi / dim1_stride (unused for 2D)
  g1[7] = 0;
  v4i gz = {0, 0, 0, 0};        // groups 2/3 unused (2-D tensor)
#if __clang_major__ >= 23
  v8i gz8 = {0, 0, 0, 0, 0, 0, 0, 0};
  __builtin_amdgcn_tensor_load_to_lds(g0, g1, gz, gz, gz8, 0);
#else
  __builtin_amdgcn_tensor_load_to_lds(g0, g1, gz, gz, 0);
#endif
}

// ---------------------------------------------------------------------------
// Phase 2: 32768 serial LSTM steps. 64 persistent WGs; WG owns 8 dims
// (32 gate rows). Wh slice DMA'd into LDS by the TDM (bf16, padded stride).
// Cross-WG h broadcast via the hs output region + L2 atomic sense barrier.
// ---------------------------------------------------------------------------
__global__ __launch_bounds__(256) void lstm_recurrence(
    const unsigned short* __restrict__ WHBF, const float* __restrict__ h0,
    const float* __restrict__ c0, const float* __restrict__ ZX,
    float* __restrict__ HS, int* barrier_cnt, int* barrier_sense) {
  __shared__ unsigned short Whs[RPB * WSTR];
  __shared__ float hsh[DIM];
  __shared__ float zrow[RPB];
  __shared__ float cs[DPB];

  const int tid = threadIdx.x;
  const int blk = blockIdx.x;

  // one-shot TDM load of this WG's Wh slice (32 KB) into LDS; wave 0 issues,
  // EXEC is ignored by tensor ops so the lane predicate just selects the wave.
  if (tid == 0) {
    unsigned lds_off = (unsigned)(unsigned long long)(lds_ushort*)Whs;
    tdm_load_wh(WHBF + (size_t)blk * RPB * DIM, lds_off);
    __builtin_amdgcn_s_wait_tensorcnt(0);
  }
  if (tid < DPB) cs[tid] = c0[blk * DPB + tid];
  __syncthreads();

  const int j = tid >> 3;                 // owned row 0..31
  const int sub = tid & 7;                // 8 lanes per row
  const int grow = (j >> 3) * DIM + blk * DPB + (j & 7);

  for (int t = 0; t < BT; ++t) {
    if (t > 0) {
      // device-wide barrier: make HS[t-1] from all 64 WGs visible
      __syncthreads();
      if (tid == 0) {
        __threadfence();
        if (atomicAdd(barrier_cnt, 1) == RB - 1) {
          atomicExch(barrier_cnt, 0);
          atomicExch(barrier_sense, t);
        } else {
          while (atomicAdd(barrier_sense, 0) < t)
            __builtin_amdgcn_s_sleep(1);
        }
      }
      __syncthreads();
    }
    const float* hsrc = (t == 0) ? h0 : &HS[(size_t)(t - 1) * DIM];
    hsh[tid] = hsrc[tid];
    hsh[tid + 256] = hsrc[tid + 256];
    __syncthreads();

    // partial matvec: row j, 64 columns per lane, rotated to avoid conflicts
    float sum = 0.f;
    const unsigned short* wr = &Whs[j * WSTR + sub * 64];
    const float* hp = &hsh[sub * 64];
#pragma unroll 8
    for (int kk = 0; kk < 64; kk += 2) {
      int c = (kk + sub * 8) & 63;
      unsigned int p = *(const unsigned int*)&wr[c];
      sum = fmaf(bf2f((unsigned short)(p & 0xffffu)), hp[c], sum);
      sum = fmaf(bf2f((unsigned short)(p >> 16)),     hp[c + 1], sum);
    }
    sum += __shfl_xor(sum, 4, 32);
    sum += __shfl_xor(sum, 2, 32);
    sum += __shfl_xor(sum, 1, 32);
    if (sub == 0) zrow[j] = sum + ZX[(size_t)t * NROWS + grow];
    __syncthreads();

    if (tid < DPB) {
      float f  = 1.f / (1.f + __expf(-zrow[tid]));
      float o  = 1.f / (1.f + __expf(-zrow[8 + tid]));
      float ii = 1.f / (1.f + __expf(-zrow[16 + tid]));
      float g  = tanhf(zrow[24 + tid]);
      float c  = f * cs[tid] + ii * g;
      cs[tid] = c;
      HS[(size_t)t * DIM + blk * DPB + tid] = o * tanhf(c);
      __threadfence();
    }
    __syncthreads();
  }
}

// ---------------------------------------------------------------------------
// Phase 3: logits = h_last @ proj_w^T + proj_b ; softmax. One block per batch.
// ---------------------------------------------------------------------------
__global__ __launch_bounds__(256) void proj_softmax(
    const float* __restrict__ HS, const float* __restrict__ PW,
    const float* __restrict__ PB, float* __restrict__ OUT) {
  __shared__ float hl[DIM];
  __shared__ float lg[OUT_SZ];
  __shared__ float red[256];
  const int b = blockIdx.x, tid = threadIdx.x;
  const float* hrow = &HS[((size_t)b * TLEN + (TLEN - 1)) * DIM];
  hl[tid] = hrow[tid];
  hl[tid + 256] = hrow[tid + 256];
  __syncthreads();

  for (int n = tid; n < OUT_SZ; n += 256) {
    float s = PB[n];
    const float* w = &PW[(size_t)n * DIM];
#pragma unroll 4
    for (int k = 0; k < DIM; ++k) s = fmaf(w[k], hl[k], s);
    lg[n] = s;
  }
  __syncthreads();

  float m = -1e30f;
  for (int n = tid; n < OUT_SZ; n += 256) m = fmaxf(m, lg[n]);
  red[tid] = m; __syncthreads();
  for (int s = 128; s > 0; s >>= 1) {
    if (tid < s) red[tid] = fmaxf(red[tid], red[tid + s]);
    __syncthreads();
  }
  m = red[0]; __syncthreads();

  float lsum = 0.f;
  for (int n = tid; n < OUT_SZ; n += 256) {
    float e = __expf(lg[n] - m);
    lg[n] = e;
    lsum += e;
  }
  red[tid] = lsum; __syncthreads();
  for (int s = 128; s > 0; s >>= 1) {
    if (tid < s) red[tid] += red[tid + s];
    __syncthreads();
  }
  const float inv = 1.f / red[0];
  for (int n = tid; n < OUT_SZ; n += 256) OUT[(size_t)b * OUT_SZ + n] = lg[n] * inv;
}

__global__ void init_barrier(int* p) {
  if (threadIdx.x < 128) p[threadIdx.x] = 0;   // cnt @ p[0], sense @ p[64]
}

// ---------------------------------------------------------------------------
extern "C" void kernel_launch(void* const* d_in, const int* in_sizes, int n_in,
                              void* d_out, int out_size, void* d_ws, size_t ws_size,
                              hipStream_t stream) {
  (void)in_sizes; (void)n_in; (void)out_size; (void)ws_size;
  const float* x  = (const float*)d_in[0];
  const float* W  = (const float*)d_in[1];
  const float* b  = (const float*)d_in[2];
  const float* h0 = (const float*)d_in[3];
  const float* c0 = (const float*)d_in[4];
  const float* pw = (const float*)d_in[5];
  const float* pb = (const float*)d_in[6];

  float* out = (float*)d_out;                 // [64*1000] softmax probs
  float* hs  = out + (size_t)BATCH * OUT_SZ;  // [BT*512] hidden states

  // workspace: zx | barrier | block-major bf16 Wh
  float* zx = (float*)d_ws;                              // BT*2048 f32
  char*  base = (char*)d_ws + (size_t)BT * NROWS * sizeof(float);
  int*   bar  = (int*)base;                              // 128 ints (1 KB pad)
  unsigned short* whbf = (unsigned short*)(base + 1024); // 2048*512 bf16

  init_barrier<<<1, 128, 0, stream>>>(bar);
  wh_to_bf16<<<(NROWS * DIM) / 256, 256, 0, stream>>>(W, whbf);
  dim3 g1(NROWS / BN, BT / BM);               // 32 x 256 blocks
  zx_gemm_bf16_wmma<<<g1, 256, 0, stream>>>(x, W, b, zx);
  lstm_recurrence<<<RB, 256, 0, stream>>>(whbf, h0, c0, zx, hs, bar, bar + 64);
  proj_softmax<<<BATCH, 256, 0, stream>>>(hs, pw, pb, out);
}
